// DiGCN_IB_Sum_26465588478354
// MI455X (gfx1250) — matrix operand order repaired and checked by reading, not compiled, not run on hardware
//
#include <hip/hip_runtime.h>
#include <hip/hip_bf16.h>

typedef __attribute__((ext_vector_type(16))) __bf16 v16bf;
typedef __attribute__((ext_vector_type(8)))  float  v8f;

__device__ __forceinline__ unsigned short f2bf(float f) {
    __bf16 b = (__bf16)f;                       // native cvt (RNE)
    return __builtin_bit_cast(unsigned short, b);
}
__device__ __forceinline__ float bf2f(unsigned short s) {
    return __uint_as_float(((unsigned)s) << 16);
}

// Transpose + convert one 128x128 fp32 weight (row-major [k][n]) to bf16 [n][k].
__global__ __launch_bounds__(256) void convw_kernel(const float* __restrict__ w,
                                                    unsigned short* __restrict__ wT) {
    int idx = blockIdx.x * 256 + threadIdx.x;   // 16384 total
    int k = idx >> 7;
    int n = idx & 127;
    wT[(size_t)n * 128 + k] = f2bf(w[(size_t)k * 128 + n]);
}

// bias3[c] = ln_b[c] + c1_b[c] + c2_b[c]
__global__ void bias3_kernel(const float* __restrict__ b0, const float* __restrict__ b1,
                             const float* __restrict__ b2, float* __restrict__ out) {
    int i = threadIdx.x;
    out[i] = b0[i] + b1[i] + b2[i];
}

// fp32 -> bf16 bulk convert of the layer input (4 elems/thread).
__global__ __launch_bounds__(256) void cvtx_kernel(const float* __restrict__ x,
                                                   unsigned short* __restrict__ xb,
                                                   int n4) {
    int i = blockIdx.x * 256 + threadIdx.x;
    if (i >= n4) return;
    const float4 v = ((const float4*)x)[i];
    ushort4 o;
    o.x = f2bf(v.x); o.y = f2bf(v.y); o.z = f2bf(v.z); o.w = f2bf(v.w);
    ((ushort4*)xb)[i] = o;
}

// Fused triple GEMM: block = 192 threads = 6 waves = 2 row-groups x 3 matrices.
// Each wave: 16 rows x 128 cols of one output.
//   mat 0: out0 = x @ ln_w + bias3   (fp32 accumulator buffer, scatter target)
//   mat 1: h1   = bf16(x @ c1_w)
//   mat 2: h2   = bf16(x @ c2_w)
// A is pre-converted bf16 (xb); all 8 B fragments of a K-step are preloaded so
// the 16 b128 loads clause together and overlap the WMMA pipeline.
__global__ __launch_bounds__(192) void gemm3_kernel(
    const unsigned short* __restrict__ xb,      // bf16 [N][128]
    const unsigned short* __restrict__ wT_ln,   // bf16 [n][k]
    const unsigned short* __restrict__ wT_c1,
    const unsigned short* __restrict__ wT_c2,
    const float* __restrict__ bias3,
    float* __restrict__ out0,
    unsigned short* __restrict__ h1,
    unsigned short* __restrict__ h2,
    int N)
{
    const int wv   = threadIdx.x >> 5;
    const int lane = threadIdx.x & 31;
    const int m    = lane & 15;     // A: M index / B,C: N index
    const int hh   = lane >> 4;     // half-wave select
    const int mat  = wv % 3;
    const int rg   = wv / 3;        // row-group 0/1
    const int row0 = blockIdx.x * 32 + rg * 16;

    const unsigned short* wsel = (mat == 0) ? wT_ln : (mat == 1) ? wT_c1 : wT_c2;

    v8f acc[8];
    #pragma unroll
    for (int i = 0; i < 8; ++i) acc[i] = (v8f){0.f,0.f,0.f,0.f,0.f,0.f,0.f,0.f};

    int arow = row0 + m;
    if (arow >= N) arow = N - 1;
    const unsigned short* xr = xb + (size_t)arow * 128;
    const unsigned short* wr = wsel + (size_t)m * 128;

    #pragma unroll
    for (int kb = 0; kb < 128; kb += 32) {
        // A fragment: lane m holds K = hh*16+{0..15} -> two 16B loads
        union { v16bf v; uint4 q[2]; } A;
        A.q[0] = *(const uint4*)(const void*)(xr + kb + hh * 16);
        A.q[1] = *(const uint4*)(const void*)(xr + kb + hh * 16 + 8);

        // Preload all 8 B fragments: lane: N = m, K = kb + hh*16 + j (32B contiguous)
        v16bf Bf[8];
        #pragma unroll
        for (int ct = 0; ct < 8; ++ct)
            Bf[ct] = *(const v16bf*)(const void*)(wr + (size_t)ct * 16 * 128 + kb + hh * 16);

        #pragma unroll
        for (int ct = 0; ct < 8; ++ct)
            acc[ct] = __builtin_amdgcn_wmma_f32_16x16x32_bf16(
                false, A.v, false, Bf[ct], (short)0, acc[ct], false, false);
    }

    // C/D layout: acc[ct][r] -> row = row0 + hh*8 + r, col = ct*16 + m
    if (mat == 0) {
        #pragma unroll
        for (int ct = 0; ct < 8; ++ct) {
            const float b = bias3[ct * 16 + m];
            #pragma unroll
            for (int r = 0; r < 8; ++r) {
                const int row = row0 + hh * 8 + r;
                if (row < N) out0[(size_t)row * 128 + ct * 16 + m] = acc[ct][r] + b;
            }
        }
    } else {
        unsigned short* ho = (mat == 1) ? h1 : h2;
        #pragma unroll
        for (int ct = 0; ct < 8; ++ct) {
            #pragma unroll
            for (int r = 0; r < 8; ++r) {
                const int row = row0 + hh * 8 + r;
                if (row < N) ho[(size_t)row * 128 + ct * 16 + m] = f2bf(acc[ct][r]);
            }
        }
    }
}

// One wave per edge: lane l handles features [4l, 4l+4).
// Wave-uniform edge index (readfirstlane) -> scalar-friendly src/dst/weight loads;
// 256B coalesced bf16 gather + 4 hardware f32 atomic adds per lane.
__global__ __launch_bounds__(256) void scatter_add_kernel(
    const int* __restrict__ src, const int* __restrict__ dst,
    const float* __restrict__ ea,
    const unsigned short* __restrict__ h,
    float* __restrict__ out, int E)
{
    const int gw   = (blockIdx.x * 256 + threadIdx.x) >> 5;
    const int lane = threadIdx.x & 31;
    if (gw >= E) return;
    const int e = __builtin_amdgcn_readfirstlane(gw);
    const int   s = src[e];
    const int   d = dst[e];
    const float w = ea[e];
    const ushort4 hv = *(const ushort4*)(h + (size_t)s * 128 + lane * 4);
    float* o = out + (size_t)d * 128 + lane * 4;
    unsafeAtomicAdd(o + 0, bf2f(hv.x) * w);
    unsafeAtomicAdd(o + 1, bf2f(hv.y) * w);
    unsafeAtomicAdd(o + 2, bf2f(hv.z) * w);
    unsafeAtomicAdd(o + 3, bf2f(hv.w) * w);
}

extern "C" void kernel_launch(void* const* d_in, const int* in_sizes, int n_in,
                              void* d_out, int out_size, void* d_ws, size_t ws_size,
                              hipStream_t stream) {
    const float* x   = (const float*)d_in[0];
    const int*   ei1 = (const int*)  d_in[1];
    const float* ea1 = (const float*)d_in[2];
    const int*   ei2 = (const int*)  d_in[3];
    const float* ea2 = (const float*)d_in[4];

    const int N  = in_sizes[0] / 128;
    const int E1 = in_sizes[1] / 2;
    const int E2 = in_sizes[3] / 2;

    // workspace layout
    char* ws = (char*)d_ws;
    size_t off = 0;
    unsigned short* wT = (unsigned short*)(ws + off); off += (size_t)9 * 16384 * 2;
    float* bias3 = (float*)(ws + off);                off += (size_t)3 * 128 * 4;
    off = (off + 255) & ~(size_t)255;
    unsigned short* h1 = (unsigned short*)(ws + off); off += (size_t)N * 128 * 2;
    off = (off + 255) & ~(size_t)255;
    unsigned short* h2 = (unsigned short*)(ws + off); off += (size_t)N * 128 * 2;
    off = (off + 255) & ~(size_t)255;
    unsigned short* xb = (unsigned short*)(ws + off); off += (size_t)N * 128 * 2;
    off = (off + 255) & ~(size_t)255;
    float* xA = (float*)(ws + off);                   off += (size_t)N * 128 * 4;

    // pre-transpose + bf16-convert all 9 weight matrices; fuse the 3 biases per layer
    for (int mat = 0; mat < 9; ++mat) {
        const int layer = mat / 3, p = mat % 3;
        const float* w = (const float*)d_in[5 + layer * 6 + p * 2];
        convw_kernel<<<64, 256, 0, stream>>>(w, wT + (size_t)mat * 16384);
    }
    for (int layer = 0; layer < 3; ++layer) {
        bias3_kernel<<<1, 128, 0, stream>>>(
            (const float*)d_in[5 + layer * 6 + 1],
            (const float*)d_in[5 + layer * 6 + 3],
            (const float*)d_in[5 + layer * 6 + 5],
            bias3 + layer * 128);
    }

    const int n4           = (N * 128) / 4;
    const int cvt_blocks   = (n4 + 255) / 256;
    const int gemm_blocks  = (N + 31) / 32;
    const int scat_blocks1 = (E1 + 7) / 8;   // 8 waves / 256-thread block
    const int scat_blocks2 = (E2 + 7) / 8;

    const float* xin = x;
    for (int layer = 0; layer < 3; ++layer) {
        // layer 0/1 accumulate into xA (reused: gemm fully rewrites it and only
        // reads xb, which was converted before), layer 2 into d_out.
        float* out = (layer == 2) ? (float*)d_out : xA;
        const unsigned short* wl = wT + (size_t)(layer * 3 + 0) * 16384;
        const unsigned short* w1 = wT + (size_t)(layer * 3 + 1) * 16384;
        const unsigned short* w2 = wT + (size_t)(layer * 3 + 2) * 16384;

        cvtx_kernel<<<cvt_blocks, 256, 0, stream>>>(xin, xb, n4);

        gemm3_kernel<<<gemm_blocks, 192, 0, stream>>>(
            xb, wl, w1, w2, bias3 + layer * 128, out, h1, h2, N);

        scatter_add_kernel<<<scat_blocks1, 256, 0, stream>>>(
            ei1, ei1 + E1, ea1, h1, out, E1);
        scatter_add_kernel<<<scat_blocks2, 256, 0, stream>>>(
            ei2, ei2 + E2, ea2, h2, out, E2);

        xin = out;
    }
}
